// Vocos_35493609734425
// MI455X (gfx1250) — matrix operand order repaired and checked by reading, not compiled, not run on hardware
//
#include <hip/hip_runtime.h>
#include <hip/hip_bf16.h>
#include <math.h>

typedef __attribute__((ext_vector_type(16))) __bf16 v16bf;
typedef __attribute__((ext_vector_type(8)))  __bf16 v8bf;
typedef __attribute__((ext_vector_type(8)))  float  v8f;

#define MROWS 32768   // B*T = 16*2048
#define K1    512     // D
#define NPAD  1088    // 1026 padded to multiple of 64
#define NFFT  1024
#define NOUT  524288  // (T-1)*HOP + WIN - 2*pad
#define TWO_PI 6.28318530717958647692f

// ---------------- elementwise helpers ----------------

__global__ __launch_bounds__(256) void k_f32_to_bf16(const float* __restrict__ in,
                                                     __bf16* __restrict__ out, int n) {
    int i = blockIdx.x * 256 + threadIdx.x;
    if (i < n) out[i] = (__bf16)in[i];
}

// W: (512 x 1026) fp32 row-major  ->  WT: (1088 x 512) bf16 row-major (K-major)
__global__ __launch_bounds__(256) void k_transpose_w(const float* __restrict__ W,
                                                     __bf16* __restrict__ WT) {
    int idx = blockIdx.x * 256 + threadIdx.x;      // over 1088*512
    int o = idx >> 9, d = idx & 511;
    WT[idx] = (o < 1026) ? (__bf16)W[d * 1026 + o] : (__bf16)0.0f;
}

// h: (M x 1088) fp32 -> packed S: (M x 1088) bf16
// cols [0,513): Re = min(exp(mag),100)*cos(p); cols [513,1026): Im = mag*sin(p); pad 0
__global__ __launch_bounds__(256) void k_make_spack(const float* __restrict__ h,
                                                    __bf16* __restrict__ S) {
    long idx = (long)blockIdx.x * 256 + threadIdx.x;   // over M*1088
    int c = (int)(idx % NPAD);
    long r = idx / NPAD;
    const float* hr = h + r * NPAD;
    float v = 0.0f;
    if (c < 513) {
        float mag = fminf(__expf(hr[c]), 100.0f);
        v = mag * __cosf(hr[c + 513]);
    } else if (c < 1026) {
        int k = c - 513;
        float mag = fminf(__expf(hr[k]), 100.0f);
        v = mag * __sinf(hr[c]);
    }
    S[idx] = (__bf16)v;
}

// BasisT[n][k], n in [0,1024), k in [0,1088), bf16, K-major rows of length 1088.
// Folds Hann window and the 1/N irfft normalization. Exact angle reduction via (k*n)&1023.
__global__ __launch_bounds__(256) void k_build_basisT(__bf16* __restrict__ BT) {
    int idx = blockIdx.x * 256 + threadIdx.x;      // over 1024*1088
    int n = idx / NPAD, k = idx % NPAD;
    float hann = 0.5f - 0.5f * __cosf((float)n * (TWO_PI / 1023.0f));
    float v = 0.0f;
    if (k < 513) {
        float w = (k == 0 || k == 512) ? 1.0f : 2.0f;
        int a = (k * n) & (NFFT - 1);
        v = w * __cosf((float)a * (TWO_PI / NFFT)) * hann * (1.0f / NFFT);
    } else if (k < 1026) {
        int kk = k - 513;
        float w = (kk == 0 || kk == 512) ? 1.0f : 2.0f;
        int a = (kk * n) & (NFFT - 1);
        v = -w * __sinf((float)a * (TWO_PI / NFFT)) * hann * (1.0f / NFFT);
    }
    BT[idx] = (__bf16)v;
}

// ---------------- bf16 WMMA GEMM ----------------
// Fragment gathers follow the ISA 16-bit A / B VGPR layouts exactly.

__device__ __forceinline__ v16bf load_afrag(const __bf16* __restrict__ A,
                                            int row, int K, int k0, int lh) {
    // lane holds row; K chunks [k0+lh*8, +8) and [k0+16+lh*8, +8)
    const __bf16* p = A + (size_t)row * K + k0 + lh * 8;
    v8bf lo = *(const v8bf*)p;
    v8bf hi = *(const v8bf*)(p + 16);
    v16bf r;
#pragma unroll
    for (int e = 0; e < 8; ++e) { r[e] = lo[e]; r[8 + e] = hi[e]; }
    return r;
}

__device__ __forceinline__ v16bf load_bfrag(const __bf16* __restrict__ BT,
                                            int col, int K, int k0, int lh) {
    // lane holds col; K chunks [k0+lh*8, +8) and [k0+16+lh*8, +8) (BT rows are K-major)
    const __bf16* p = BT + (size_t)col * K + k0 + lh * 8;
    v8bf lo = *(const v8bf*)p;
    v8bf hi = *(const v8bf*)(p + 16);
    v16bf r;
#pragma unroll
    for (int e = 0; e < 8; ++e) { r[e] = lo[e]; r[8 + e] = hi[e]; }
    return r;
}

// C(MxN,f32) = A(MxK,bf16 row-major) * B, BT = B transposed (N x K, bf16 row-major).
// One wave computes a 64x64 tile: 4 M-frags x 4 N-frags, K stepped by 32,
// register double-buffered with the K-loop unrolled by 64 (K % 64 == 0 required).
__global__ __launch_bounds__(256)
void k_gemm_bf16(const __bf16* __restrict__ A, const __bf16* __restrict__ BT,
                 float* __restrict__ C, const float* __restrict__ bias,
                 int M, int N, int K, int biasN) {
    int wave = (blockIdx.x * 256 + threadIdx.x) >> 5;
    int lane = threadIdx.x & 31;
    int nT = N >> 6, mT = M >> 6;
    if (wave >= mT * nT) return;
    int mt = wave / nT, nt = wave % nT;
    int lh  = lane >> 4;   // lane half: 0 or 1
    int l16 = lane & 15;
    int rowBase = mt * 64 + l16;
    int colBase = nt * 64 + l16;

    v8f acc[4][4] = {};
    v16bf a0[4], b0[4], a1[4], b1[4];

#pragma unroll
    for (int i = 0; i < 4; ++i) a0[i] = load_afrag(A, rowBase + i * 16, K, 0, lh);
#pragma unroll
    for (int j = 0; j < 4; ++j) b0[j] = load_bfrag(BT, colBase + j * 16, K, 0, lh);

    for (int k0 = 0; k0 < K; k0 += 64) {
        // prefetch k0+32 while computing k0
#pragma unroll
        for (int i = 0; i < 4; ++i) a1[i] = load_afrag(A, rowBase + i * 16, K, k0 + 32, lh);
#pragma unroll
        for (int j = 0; j < 4; ++j) b1[j] = load_bfrag(BT, colBase + j * 16, K, k0 + 32, lh);
#pragma unroll
        for (int i = 0; i < 4; ++i)
#pragma unroll
            for (int j = 0; j < 4; ++j)
                acc[i][j] = __builtin_amdgcn_wmma_f32_16x16x32_bf16(
                    false, a0[i], false, b0[j], (short)0, acc[i][j], false, false);
        if (k0 + 64 < K) {
#pragma unroll
            for (int i = 0; i < 4; ++i) a0[i] = load_afrag(A, rowBase + i * 16, K, k0 + 64, lh);
#pragma unroll
            for (int j = 0; j < 4; ++j) b0[j] = load_bfrag(BT, colBase + j * 16, K, k0 + 64, lh);
        }
#pragma unroll
        for (int i = 0; i < 4; ++i)
#pragma unroll
            for (int j = 0; j < 4; ++j)
                acc[i][j] = __builtin_amdgcn_wmma_f32_16x16x32_bf16(
                    false, a1[i], false, b1[j], (short)0, acc[i][j], false, false);
    }

    // Epilogue: f32 C/D layout -> element e: row = m0 + lh*8 + e, col = n0 + l16
#pragma unroll
    for (int i = 0; i < 4; ++i)
#pragma unroll
        for (int j = 0; j < 4; ++j) {
            int col = nt * 64 + j * 16 + l16;
            float bv = (bias != nullptr && col < biasN) ? bias[col] : 0.0f;
#pragma unroll
            for (int e = 0; e < 8; ++e) {
                int row = mt * 64 + i * 16 + lh * 8 + e;
                C[(size_t)row * N + col] = acc[i][j][e] + bv;
            }
        }
}

// ---------------- overlap-add gather + envelope + trim ----------------
// frames: (32768 x 1024) fp32 (already windowed & /N). out: (16 x 524288).
__global__ __launch_bounds__(256) void k_ola(const float* __restrict__ frames,
                                             float* __restrict__ out) {
    long idx = (long)blockIdx.x * 256 + threadIdx.x;   // over 16*524288
    int b = (int)(idx >> 19);
    int j = (int)(idx & (NOUT - 1));
    int m = j + 384;                                   // un-trimmed sample index
    int thi = m >> 8;            if (thi > 2047) thi = 2047;
    int tlo = (m - 768) >> 8;    if (tlo < 0)    tlo = 0;   // ceil((m-1023)/256)
    float y = 0.0f, env = 0.0f;
    for (int t = tlo; t <= thi; ++t) {
        int n = m - (t << 8);
        y += frames[((size_t)(b * 2048 + t) << 10) + n];
        float w = 0.5f - 0.5f * __cosf((float)n * (TWO_PI / 1023.0f));
        env += w * w;
    }
    out[idx] = y / (env + 1e-11f);
}

// ---------------- launch ----------------
extern "C" void kernel_launch(void* const* d_in, const int* in_sizes, int n_in,
                              void* d_out, int out_size, void* d_ws, size_t ws_size,
                              hipStream_t stream) {
    const float* x = (const float*)d_in[0];   // (16,2048,512)
    const float* W = (const float*)d_in[1];   // (512,1026)
    const float* b = (const float*)d_in[2];   // (1026,)
    float* out = (float*)d_out;               // (16,524288)
    char* ws = (char*)d_ws;

    size_t off = 0;
    __bf16* A1 = (__bf16*)(ws + off); off += (size_t)MROWS * K1 * 2;     //  33.6 MB
    __bf16* WT = (__bf16*)(ws + off); off += (size_t)NPAD * K1 * 2;      //   1.1 MB
    __bf16* Sp = (__bf16*)(ws + off); off += (size_t)MROWS * NPAD * 2;   //  71.3 MB
    __bf16* BT = (__bf16*)(ws + off); off += (size_t)NFFT * NPAD * 2;    //   2.2 MB
    float*  H  = (float*)(ws + off);                                     // 142.6 MB
    float*  frames = H;  // reuse: frames (M x 1024 f32) fits in H (M x 1088 f32)

    // 1) x -> bf16
    k_f32_to_bf16<<<(MROWS * K1) / 256, 256, 0, stream>>>(x, A1, MROWS * K1);
    // 2) W -> WT (padded, K-major)
    k_transpose_w<<<(NPAD * K1) / 256, 256, 0, stream>>>(W, WT);
    // 3) GEMM1: H = A1 * W + b   (32768 x 1088, cols >=1026 are zero)
    {
        int waves = (MROWS / 64) * (NPAD / 64);        // 512*17 = 8704
        k_gemm_bf16<<<(waves * 32) / 256, 256, 0, stream>>>(A1, WT, H, b,
                                                            MROWS, NPAD, K1, 1026);
    }
    // 4) packed spectrum (bf16)
    k_make_spack<<<(int)(((long)MROWS * NPAD) / 256), 256, 0, stream>>>(H, Sp);
    // 5) windowed/normalized DFT basis (transposed)
    k_build_basisT<<<(NFFT * NPAD) / 256, 256, 0, stream>>>(BT);
    // 6) GEMM2: frames = Sp * Basis   (32768 x 1024)
    {
        int waves = (MROWS / 64) * (NFFT / 64);        // 512*16 = 8192
        k_gemm_bf16<<<(waves * 32) / 256, 256, 0, stream>>>(Sp, BT, frames, nullptr,
                                                            MROWS, NFFT, NPAD, 0);
    }
    // 7) overlap-add gather + envelope division + trim
    k_ola<<<(16 * NOUT) / 256, 256, 0, stream>>>(frames, out);
}
